// LSTransformerHead_64836826300855
// MI455X (gfx1250) — compile-verified, matrix operand order checked by reading, not run on hardware
//
#include <hip/hip_runtime.h>

// ---------------- problem constants (match reference) ----------------
#define B_      8
#define N_      4096
#define D_      512
#define H_      8
#define DH_     64
#define R_      256
#define DEPTH_  4
#define NC_     10
#define BN_     (B_ * N_)      // 32768 token rows
#define BH_     (B_ * H_)      // 64 (b,h) batches

// ---------------- WMMA types ----------------
typedef __attribute__((ext_vector_type(16))) __bf16  v16bf;
typedef __attribute__((ext_vector_type(8)))  float   v8f;
typedef __attribute__((ext_vector_type(4)))  unsigned int uvec4;

__device__ __forceinline__ unsigned short f2bf(float f) {
    union { float f; unsigned u; } x; x.f = f;
    unsigned u = x.u;
    unsigned r = u + 0x7FFFu + ((u >> 16) & 1u);   // round-to-nearest-even
    return (unsigned short)(r >> 16);
}

__device__ __forceinline__ float gelu_tanh(float x) {
    const float c = 0.7978845608028654f;           // sqrt(2/pi)
    float x3 = x * x * x;
    return 0.5f * x * (1.0f + tanhf(c * (x + 0.044715f * x3)));
}

// ====================================================================
// Generic batched strided GEMM:  C = alpha * A @ B  (+bias)(+gelu)(+res)
//   A[m,k] = Aptr[m*ldaM + k*ldaK], B[k,n] = Bptr[k*ldbK + n*ldbN]
//   batch z in grid.z, zb = z/bdiv, zh = z%bdiv with 2 strides each.
//   mode: 0 none | 1 +bias | 2 +bias,gelu | 3 +res | 4 +bias,+res
// Block: 256 thr = 8 waves. Tile 128x64, each wave: 16 rows x 64 cols,
// 4 accumulators -> 4 WMMAs per 32-wide k-step.
// NOTE: all launches guarantee M%128==0, N%64==0, K%32==0 (no guards).
// ====================================================================
#define BMT 128
#define BNT 64
#define BKT 32
#define LDSP 48   // padded row stride (ushorts): 96B, keeps 16B-aligned chunks

__global__ void __launch_bounds__(256)
wmma_gemm(const float* __restrict__ A, const float* __restrict__ B, float* C,
          int M, int N, int K,
          long ldaM, long ldaK, long ldbK, long ldbN, long ldcM,
          int bdiv, long sA1, long sA2, long sB1, long sB2, long sC1, long sC2,
          float alpha, int mode,
          const float* __restrict__ bias,
          const float* res, long ldresM, long sR1, long sR2)
{
    int z  = blockIdx.z;
    int zb = z / bdiv, zh = z % bdiv;
    A += (long)zb * sA1 + (long)zh * sA2;
    B += (long)zb * sB1 + (long)zh * sB2;
    C += (long)zb * sC1 + (long)zh * sC2;
    if (res) res += (long)zb * sR1 + (long)zh * sR2;

    __shared__ __align__(16) unsigned short As[BMT][LDSP];
    __shared__ __align__(16) unsigned short Bs[BNT][LDSP];   // transposed: [n][k]

    int tid  = threadIdx.x;
    int lane = tid & 31;
    int wave = tid >> 5;      // 0..7 -> m-strip
    int hlf  = lane >> 4;
    int lm   = lane & 15;

    int m0 = blockIdx.y * BMT;
    int n0 = blockIdx.x * BNT;

    v8f acc[4] = {{}, {}, {}, {}};

    for (int k0 = 0; k0 < K; k0 += BKT) {
        // ---- stage A tile (128x32) as packed bf16 pairs along k ----
        // 2048 pairs, 8 per thread; pair p: row = p>>4, col = (p&15)*2
        #pragma unroll
        for (int j = 0; j < 8; ++j) {
            int p = tid + j * 256;
            int r = p >> 4;
            int c = (p & 15) << 1;
            const float* ap = A + (long)(m0 + r) * ldaM + (long)(k0 + c) * ldaK;
            unsigned pk = (unsigned)f2bf(ap[0]) | ((unsigned)f2bf(ap[ldaK]) << 16);
            *(unsigned*)&As[r][c] = pk;
        }
        // ---- stage B tile (32x64) -> Bs[n][k], pairs along n ----
        // 1024 pairs, 4 per thread; pair p: k = p>>5, n = (p&31)*2
        #pragma unroll
        for (int j = 0; j < 4; ++j) {
            int p = tid + j * 256;
            int ck = p >> 5;
            int cn = (p & 31) << 1;
            const float* bp = B + (long)(k0 + ck) * ldbK + (long)(n0 + cn) * ldbN;
            Bs[cn][ck]     = f2bf(bp[0]);
            Bs[cn + 1][ck] = f2bf(bp[ldbN]);
        }
        __syncthreads();

        // ---- fragments (ISA 16-bit 16x32 layout: K chunks half*8, 16+half*8) ----
        union Frag { uvec4 q[2]; v16bf v; } a, b;
        int am = wave * 16 + lm;
        a.q[0] = *(const uvec4*)&As[am][hlf * 8];
        a.q[1] = *(const uvec4*)&As[am][16 + hlf * 8];

        #pragma unroll
        for (int t = 0; t < 4; ++t) {
            int bn = t * 16 + lm;
            b.q[0] = *(const uvec4*)&Bs[bn][hlf * 8];
            b.q[1] = *(const uvec4*)&Bs[bn][16 + hlf * 8];
            acc[t] = __builtin_amdgcn_wmma_f32_16x16x32_bf16(false, a.v, false, b.v,
                                                             (short)0, acc[t], false, false);
        }
        __syncthreads();
    }

    // ---- epilogue (C/D layout: VGPR v -> M = v + 8*half, N = lane%16) ----
    #pragma unroll
    for (int t = 0; t < 4; ++t) {
        int n = n0 + t * 16 + lm;
        float bv = (mode == 1 || mode == 2 || mode == 4) ? bias[n] : 0.0f;
        #pragma unroll
        for (int v = 0; v < 8; ++v) {
            int m = m0 + wave * 16 + v + 8 * hlf;
            float val = alpha * acc[t][v] + bv;
            if (mode == 2) val = gelu_tanh(val);
            if (mode == 3 || mode == 4) val += res[(long)m * ldresM + n];
            C[(long)m * ldcM + n] = val;
        }
    }
}

// ====================================================================
// Elementwise / reduction kernels
// ====================================================================
__global__ void embed_kernel(const int* __restrict__ x, const float* __restrict__ tok,
                             const float* __restrict__ pos, float* __restrict__ h)
{
    int row = blockIdx.x;            // b*N + n
    int n   = row & (N_ - 1);
    int id  = x[row];
    for (int e = threadIdx.x; e < D_; e += blockDim.x)
        h[(long)row * D_ + e] = tok[(long)id * D_ + e] + pos[(long)n * D_ + e];
}

// LayerNorm over rows of length `len` (blockDim power of 2 <= 256)
__global__ void ln_kernel(const float* __restrict__ in, float* __restrict__ out,
                          const float* __restrict__ g, const float* __restrict__ b, int len)
{
    long row = blockIdx.x;
    const float* p = in + row * (long)len;
    float* q = out + row * (long)len;
    __shared__ float red[256];
    int tid = threadIdx.x, bs = blockDim.x;

    float s = 0.0f;
    for (int i = tid; i < len; i += bs) s += p[i];
    red[tid] = s; __syncthreads();
    for (int st = bs >> 1; st > 0; st >>= 1) { if (tid < st) red[tid] += red[tid + st]; __syncthreads(); }
    float mean = red[0] / (float)len; __syncthreads();

    float v = 0.0f;
    for (int i = tid; i < len; i += bs) { float c = p[i] - mean; v += c * c; }
    red[tid] = v; __syncthreads();
    for (int st = bs >> 1; st > 0; st >>= 1) { if (tid < st) red[tid] += red[tid + st]; __syncthreads(); }
    float rstd = rsqrtf(red[0] / (float)len + 1e-5f); __syncthreads();

    for (int i = tid; i < len; i += bs)
        q[i] = (p[i] - mean) * rstd * g[i] + b[i];
}

// row softmax in place (rows contiguous, length len)
__global__ void softmax_rows(float* __restrict__ data, int len)
{
    long row = blockIdx.x;
    float* p = data + row * (long)len;
    __shared__ float red[256];
    int tid = threadIdx.x, bs = blockDim.x;

    float mx = -3.4e38f;
    for (int i = tid; i < len; i += bs) mx = fmaxf(mx, p[i]);
    red[tid] = mx; __syncthreads();
    for (int st = bs >> 1; st > 0; st >>= 1) { if (tid < st) red[tid] = fmaxf(red[tid], red[tid + st]); __syncthreads(); }
    mx = red[0]; __syncthreads();

    float s = 0.0f;
    for (int i = tid; i < len; i += bs) { float e = __expf(p[i] - mx); p[i] = e; s += e; }
    red[tid] = s; __syncthreads();
    for (int st = bs >> 1; st > 0; st >>= 1) { if (tid < st) red[tid] += red[tid + st]; __syncthreads(); }
    float inv = 1.0f / red[0]; __syncthreads();

    for (int i = tid; i < len; i += bs) p[i] *= inv;
}

// fused: kl = LN(k_head) w/ lnl; sloc[z,n] = (q*scale) . kl    (WIN=1 local score)
__global__ void sloc_ln_dot(const float* __restrict__ qkv, const float* __restrict__ g,
                            const float* __restrict__ bt, float* __restrict__ sloc)
{
    int n = blockIdx.x, z = blockIdx.y;
    int b = z >> 3, hh = z & 7;
    int d = threadIdx.x;   // 64 threads
    __shared__ float red[64];

    long base = ((long)(b * N_ + n)) * (3 * D_) + hh * DH_ + d;
    float kd = qkv[base + D_];

    red[d] = kd; __syncthreads();
    for (int st = 32; st > 0; st >>= 1) { if (d < st) red[d] += red[d + st]; __syncthreads(); }
    float mean = red[0] * (1.0f / DH_); __syncthreads();

    float c = kd - mean;
    red[d] = c * c; __syncthreads();
    for (int st = 32; st > 0; st >>= 1) { if (d < st) red[d] += red[d + st]; __syncthreads(); }
    float rstd = rsqrtf(red[0] * (1.0f / DH_) + 1e-5f); __syncthreads();

    float kl = c * rstd * g[d] + bt[d];
    float qd = qkv[base] * 0.125f;           // DH^-0.5
    red[d] = qd * kl; __syncthreads();
    for (int st = 32; st > 0; st >>= 1) { if (d < st) red[d] += red[d + st]; __syncthreads(); }
    if (d == 0) sloc[(long)z * N_ + n] = red[0];
}

// unified 257-way softmax over [sloc | Sg row]; normalize Sg in place,
// write o_init = w0 * v_head
__global__ void softmax_combine(float* __restrict__ sg, const float* __restrict__ sloc,
                                const float* __restrict__ qkv, float* __restrict__ o)
{
    int n = blockIdx.x, z = blockIdx.y;
    int b = z >> 3, hh = z & 7;
    int tid = threadIdx.x;   // 256 == R_
    __shared__ float red[256];
    __shared__ float sh[2];  // [0]=w0, [1]=inv_sum

    float* row = sg + ((long)z * N_ + n) * R_;
    float sl = sloc[(long)z * N_ + n];

    float v = row[tid];
    red[tid] = v; __syncthreads();
    for (int st = 128; st > 0; st >>= 1) { if (tid < st) red[tid] = fmaxf(red[tid], red[tid + st]); __syncthreads(); }
    float mx = fmaxf(red[0], sl); __syncthreads();

    float e = __expf(v - mx);
    red[tid] = e; __syncthreads();
    for (int st = 128; st > 0; st >>= 1) { if (tid < st) red[tid] += red[tid + st]; __syncthreads(); }
    if (tid == 0) {
        float esl = __expf(sl - mx);
        float inv = 1.0f / (red[0] + esl);
        sh[0] = esl * inv; sh[1] = inv;
    }
    __syncthreads();

    row[tid] = e * sh[1];
    if (tid < DH_) {
        float vv = qkv[((long)(b * N_ + n)) * (3 * D_) + 2 * D_ + hh * DH_ + tid];
        o[((long)z * N_ + n) * DH_ + tid] = sh[0] * vv;
    }
}

// o[b,h,n,d] -> ocat[b,n,h*64+d]
__global__ void merge_heads(const float* __restrict__ o, float* __restrict__ ocat)
{
    long i = (long)blockIdx.x * blockDim.x + threadIdx.x;
    if (i >= (long)BH_ * N_ * DH_) return;
    int d = (int)(i & (DH_ - 1));
    long t = i >> 6;
    int n = (int)(t & (N_ - 1));
    long t2 = t >> 12;
    int hh = (int)(t2 & 7);
    int b = (int)(t2 >> 3);
    ocat[((long)(b * N_ + n)) * D_ + hh * DH_ + d] = o[i];
}

// out[b,c] = b_final[c] + sum_k logits[b,k] * W_final[k,c]   (K = N*D)
__global__ void final_kernel(const float* __restrict__ logits, const float* __restrict__ Wf,
                             const float* __restrict__ bf, float* __restrict__ out)
{
    int c = blockIdx.x, b = blockIdx.y;
    const long Kt = (long)N_ * D_;
    __shared__ float red[256];
    int tid = threadIdx.x;
    float s = 0.0f;
    for (long i = tid; i < Kt; i += blockDim.x)
        s += logits[(long)b * Kt + i] * Wf[i * NC_ + c];
    red[tid] = s; __syncthreads();
    for (int st = 128; st > 0; st >>= 1) { if (tid < st) red[tid] += red[tid + st]; __syncthreads(); }
    if (tid == 0) out[b * NC_ + c] = red[0] + bf[c];
}

// ====================================================================
// Host-side GEMM launcher
// ====================================================================
static inline void gemm(hipStream_t st, const float* A, const float* B, float* C,
                        int M, int Nn, int K,
                        long ldaM, long ldaK, long ldbK, long ldbN, long ldcM,
                        int batch, int bdiv,
                        long sA1, long sA2, long sB1, long sB2, long sC1, long sC2,
                        float alpha, int mode, const float* bias,
                        const float* res, long ldresM, long sR1, long sR2)
{
    dim3 grid(Nn / BNT, M / BMT, batch);
    wmma_gemm<<<grid, 256, 0, st>>>(A, B, C, M, Nn, K,
                                    ldaM, ldaK, ldbK, ldbN, ldcM,
                                    bdiv, sA1, sA2, sB1, sB2, sC1, sC2,
                                    alpha, mode, bias, res, ldresM, sR1, sR2);
}

extern "C" void kernel_launch(void* const* d_in, const int* in_sizes, int n_in,
                              void* d_out, int out_size, void* d_ws, size_t ws_size,
                              hipStream_t stream)
{
    (void)in_sizes; (void)n_in; (void)out_size; (void)ws_size;

    const int*   x     = (const int*)  d_in[0];
    const float* tok   = (const float*)d_in[1];
    const float* pos   = (const float*)d_in[2];
    const float* ln1_g = (const float*)d_in[3];
    const float* ln1_b = (const float*)d_in[4];
    const float* Wqkv  = (const float*)d_in[5];
    const float* Wdp   = (const float*)d_in[6];
    const float* lnl_g = (const float*)d_in[7];
    const float* lnl_b = (const float*)d_in[8];
    const float* lng_g = (const float*)d_in[9];
    const float* lng_b = (const float*)d_in[10];
    const float* Wout  = (const float*)d_in[11];
    const float* bout  = (const float*)d_in[12];
    const float* ln2_g = (const float*)d_in[13];
    const float* ln2_b = (const float*)d_in[14];
    const float* W1    = (const float*)d_in[15];
    const float* b1    = (const float*)d_in[16];
    const float* W2    = (const float*)d_in[17];
    const float* b2    = (const float*)d_in[18];
    const float* lnf_g = (const float*)d_in[19];
    const float* lnf_b = (const float*)d_in[20];
    const float* Wlog  = (const float*)d_in[21];
    const float* blog  = (const float*)d_in[22];
    const float* Wfin  = (const float*)d_in[23];
    const float* bfin  = (const float*)d_in[24];

    // ---------------- workspace layout (fp32) ----------------
    float* ws   = (float*)d_ws;
    float* h    = ws;                               // [BN, D]
    float* xn   = h    + (long)BN_ * D_;            // [BN, D]
    float* qkv  = xn   + (long)BN_ * D_;            // [BN, 3D]  (reused for logits)
    float* big  = qkv  + (long)BN_ * 3 * D_;        // [BH,R,N] / [BH,N,R] / [BN,4D]
    float* gk   = big  + (long)BH_ * N_ * R_;       // [BH, R, DH]
    float* gv   = gk   + (long)BH_ * R_ * DH_;      // [BH, R, DH]
    float* gkn  = gv   + (long)BH_ * R_ * DH_;      // [BH, R, DH]
    float* slc  = gkn  + (long)BH_ * R_ * DH_;      // [BH, N]
    float* o    = slc  + (long)BH_ * N_;            // [BH, N, DH]
    float* ocat = o    + (long)BH_ * N_ * DH_;      // [BN, D]

    const long QS = 3 * D_;                          // qkv row stride

    // h = tok_emb[x] + pos_emb
    embed_kernel<<<BN_, 256, 0, stream>>>(x, tok, pos, h);

    for (int i = 0; i < DEPTH_; ++i) {
        const float* Wqkv_i = Wqkv + (long)i * D_ * 3 * D_;
        const float* Wdp_i  = Wdp  + (long)i * DH_ * R_;
        const float* Wout_i = Wout + (long)i * D_ * D_;
        const float* W1_i   = W1   + (long)i * D_ * 4 * D_;
        const float* W2_i   = W2   + (long)i * 4 * D_ * D_;

        // xn = LN1(h)
        ln_kernel<<<BN_, 256, 0, stream>>>(h, xn, ln1_g + i * D_, ln1_b + i * D_, D_);

        // qkv = xn @ Wqkv
        gemm(stream, xn, Wqkv_i, qkv, BN_, 3 * D_, D_,
             D_, 1, 3 * D_, 1, 3 * D_,
             1, 1, 0, 0, 0, 0, 0, 0,
             1.0f, 0, nullptr, nullptr, 0, 0, 0);

        // S^T[z,r,n] = Wdp^T @ k^T    (batched over z=(b,h); Wdp shared, M=R=256)
        gemm(stream, Wdp_i, qkv + D_, big, R_, N_, DH_,
             /*A*/ 1, R_, /*B*/ 1, QS, /*C*/ N_,
             BH_, H_, 0, 0, (long)N_ * QS, DH_, (long)H_ * R_ * N_, (long)R_ * N_,
             1.0f, 0, nullptr, nullptr, 0, 0, 0);

        // pk^T: softmax over sequence (contiguous rows of length N)
        softmax_rows<<<BH_ * R_, 256, 0, stream>>>(big, N_);

        // gk = P @ k ; gv = P @ v
        gemm(stream, big, qkv + D_, gk, R_, DH_, N_,
             N_, 1, QS, 1, DH_,
             BH_, H_, (long)H_ * R_ * N_, (long)R_ * N_, (long)N_ * QS, DH_,
             (long)H_ * R_ * DH_, (long)R_ * DH_,
             1.0f, 0, nullptr, nullptr, 0, 0, 0);
        gemm(stream, big, qkv + 2 * D_, gv, R_, DH_, N_,
             N_, 1, QS, 1, DH_,
             BH_, H_, (long)H_ * R_ * N_, (long)R_ * N_, (long)N_ * QS, DH_,
             (long)H_ * R_ * DH_, (long)R_ * DH_,
             1.0f, 0, nullptr, nullptr, 0, 0, 0);

        // gkn = DualLN_global(gk)
        ln_kernel<<<BH_ * R_, 64, 0, stream>>>(gk, gkn, lng_g + i * DH_, lng_b + i * DH_, DH_);

        // sloc[z,n] = (q*scale) . LN_local(k)    (window=1 local attention)
        sloc_ln_dot<<<dim3(N_, BH_), 64, 0, stream>>>(qkv, lnl_g + i * DH_, lnl_b + i * DH_, slc);

        // Sg[z,n,r] = scale * q @ gkn^T
        gemm(stream, qkv, gkn, big, N_, R_, DH_,
             QS, 1, 1, DH_, R_,
             BH_, H_, (long)N_ * QS, DH_, (long)H_ * R_ * DH_, (long)R_ * DH_,
             (long)H_ * N_ * R_, (long)N_ * R_,
             0.125f, 0, nullptr, nullptr, 0, 0, 0);

        // unified softmax over [sloc | Sg]; o = w0*v ; Sg normalized in place
        softmax_combine<<<dim3(N_, BH_), 256, 0, stream>>>(big, slc, qkv, o);

        // o += Sg_norm @ gv
        gemm(stream, big, gv, o, N_, DH_, R_,
             R_, 1, DH_, 1, DH_,
             BH_, H_, (long)H_ * N_ * R_, (long)N_ * R_, (long)H_ * R_ * DH_, (long)R_ * DH_,
             (long)H_ * N_ * DH_, (long)N_ * DH_,
             1.0f, 3, nullptr, o, DH_, (long)H_ * N_ * DH_, (long)N_ * DH_);

        // concat heads
        merge_heads<<<(BH_ * N_ * DH_) / 256, 256, 0, stream>>>(o, ocat);

        // h = h + ocat @ Wout + bout
        gemm(stream, ocat, Wout_i, h, BN_, D_, D_,
             D_, 1, D_, 1, D_,
             1, 1, 0, 0, 0, 0, 0, 0,
             1.0f, 4, bout + i * D_, h, D_, 0, 0);

        // FFN: xn = LN2(h); big = gelu(xn@W1+b1); h = h + big@W2 + b2
        ln_kernel<<<BN_, 256, 0, stream>>>(h, xn, ln2_g + i * D_, ln2_b + i * D_, D_);
        gemm(stream, xn, W1_i, big, BN_, 4 * D_, D_,
             D_, 1, 4 * D_, 1, 4 * D_,
             1, 1, 0, 0, 0, 0, 0, 0,
             1.0f, 2, b1 + (long)i * 4 * D_, nullptr, 0, 0, 0);
        gemm(stream, big, W2_i, h, BN_, D_, 4 * D_,
             4 * D_, 1, D_, 1, D_,
             1, 1, 0, 0, 0, 0, 0, 0,
             1.0f, 4, b2 + i * D_, h, D_, 0, 0);
    }

    // final LN + logits (logits reuse qkv buffer)
    ln_kernel<<<BN_, 256, 0, stream>>>(h, xn, lnf_g, lnf_b, D_);
    gemm(stream, xn, Wlog, qkv, BN_, D_, D_,
         D_, 1, D_, 1, D_,
         1, 1, 0, 0, 0, 0, 0, 0,
         1.0f, 1, blog, nullptr, 0, 0, 0);

    // classification head
    final_kernel<<<dim3(NC_, B_), 256, 0, stream>>>(qkv, Wfin, bfin, (float*)d_out);
}
